// Encoder_4544075399463
// MI455X (gfx1250) — matrix operand order repaired and checked by reading, not compiled, not run on hardware
//
#include <hip/hip_runtime.h>
#include <hip/hip_bf16.h>
#include <stdint.h>

// Problem dims (fixed by reference)
#define B_  64
#define T_  128
#define E_  256
#define H_  512
#define G3  1536            // 3*H
#define M_  (B_ * T_)       // 8192 rows for input-projection GEMMs
#define LDPAD (H_ + 8)      // padded LDS row stride (bf16 elems): 1040B -> bank stride 4

typedef __attribute__((ext_vector_type(16))) __bf16 v16bf;
typedef __attribute__((ext_vector_type(8)))  float  v8f;

union Frag16 { uint4 q[2]; v16bf v; };

__device__ __forceinline__ unsigned short f2bf(float f) {
  union { float f; unsigned u; } c; c.f = f;
  unsigned u = c.u;
  unsigned r = (u + 0x7FFFu + ((u >> 16) & 1u)) >> 16;   // round-to-nearest-even
  return (unsigned short)r;
}
__device__ __forceinline__ float bf2f(unsigned short b) {
  union { unsigned u; float f; } c; c.u = ((unsigned)b) << 16;
  return c.f;
}

// A fragment: 16(M) x 32(K) bf16, row-major source with row stride ldk (elements).
// lane m=lane&15, half=(lane>>4): elem i -> K=half*16+i.
__device__ __forceinline__ v16bf load_a_frag(const unsigned short* __restrict__ row0,
                                             int ldk, int lane) {
  const int m = lane & 15, half = (lane >> 4) & 1;
  const unsigned short* p = row0 + (size_t)m * ldk + half * 16;
  Frag16 f;
  f.q[0] = *(const uint4*)(p);
  f.q[1] = *(const uint4*)(p + 8);
  return f.v;
}

// B fragment: 32(K) x 16(N) bf16, from a TRANSPOSED (N x K) source with row stride ldk.
// lane n=lane&15, half=(lane>>4): elem i -> K=half*16+i => one contiguous 32B read.
__device__ __forceinline__ v16bf load_b_frag(const unsigned short* colT0,
                                             int ldk, int lane) {
  const int n = lane & 15, half = (lane >> 4) & 1;
  const unsigned short* p = colT0 + (size_t)n * ldk + half * 16;
  Frag16 f;
  f.q[0] = *(const uint4*)(p);
  f.q[1] = *(const uint4*)(p + 8);
  return f.v;
}

// -------------------- embedding gather -> bf16 --------------------
__global__ void embed_kernel(const int* __restrict__ x, const float* __restrict__ emb,
                             unsigned short* __restrict__ Xe) {
  int i = blockIdx.x * blockDim.x + threadIdx.x;   // over M_*E_
  int bt = i >> 8, e = i & 255;
  int tok = x[bt];
  Xe[i] = f2bf(emb[(size_t)tok * E_ + e]);
}

// -------------------- f32 (R x C) -> bf16 (C x R) transpose-convert --------------------
__global__ void transpose_bf16_kernel(const float* __restrict__ in,
                                      unsigned short* __restrict__ out, int R, int C) {
  int i = blockIdx.x * blockDim.x + threadIdx.x;
  if (i >= R * C) return;
  int r = i / C, c = i % C;
  out[(size_t)c * R + r] = f2bf(in[i]);
}

// -------------------- xs = Xin(bf16, MxK) @ Wk + bi -> f32 (M x 1536) --------------------
// block 256 = 8 waves; WG computes 32x64 patch: wave (mi in 0..1, ni in 0..3) -> 16x16 tile.
__global__ void gemm_xs_kernel(const unsigned short* __restrict__ Xin,
                               const unsigned short* __restrict__ WkT, // (1536 x K) bf16
                               const float* __restrict__ bias,         // bi row, 1536
                               float* __restrict__ xs, int K) {
  const int tid  = threadIdx.x;
  const int lane = tid & 31, w = tid >> 5;
  const int mi = w & 1, ni = w >> 1;
  const int m0 = (blockIdx.x * 2 + mi) * 16;
  const int n0 = (blockIdx.y * 4 + ni) * 16;
  const int nL = lane & 15, half = (lane >> 4) & 1;

  const unsigned short* A0 = Xin + (size_t)m0 * K;
  const unsigned short* B0 = WkT + (size_t)n0 * K;

  v8f acc = {};
  for (int kc = 0; kc < K; kc += 32) {
    v16bf a = load_a_frag(A0 + kc, K, lane);
    v16bf b = load_b_frag(B0 + kc, K, lane);
    acc = __builtin_amdgcn_wmma_f32_16x16x32_bf16(false, a, false, b, (short)0, acc,
                                                  false, false);
  }
  const float bv = bias[n0 + nL];
#pragma unroll
  for (int r = 0; r < 8; ++r) {
    const int m = m0 + half * 8 + r;
    xs[(size_t)m * G3 + n0 + nL] = acc[r] + bv;
  }
}

// -------------------- cooperative GRU scan (one layer, both directions) --------------------
// grid.x = 32: blockIdx>>4 = dir, blockIdx&15 = 32-column slice of h.
// 8 waves = 4 M-tiles x 2 col-tiles; each wave: 3 WMMA accumulators (z,r,n gates).
// Wr slice (96 rows x 512 K bf16) is staged once into padded LDS; B fragments come from
// LDS (ds_load_b128) every step, A fragments (cross-WG h) from global.
__global__ void gru_scan_kernel(const float* __restrict__ xs0, const float* __restrict__ xs1,
                                const unsigned short* __restrict__ WrT0,
                                const unsigned short* __restrict__ WrT1,   // (1536 x 512) bf16
                                const float* __restrict__ bias0, const float* __restrict__ bias1,
                                unsigned short* __restrict__ hb0,          // 2 ping-pong bufs, 64x512 bf16 each
                                unsigned short* __restrict__ hb1,
                                float* __restrict__ hstate0, float* __restrict__ hstate1,
                                unsigned short* __restrict__ Y,            // (B*T x 1024) bf16
                                unsigned* __restrict__ bar, int nwg) {
  extern __shared__ unsigned short sWr[];    // [96][LDPAD] bf16 (~97.5 KB)

  const int tid  = threadIdx.x;
  const int lane = tid & 31, w = tid >> 5;
  const int dir  = blockIdx.x >> 4;
  const int wgi  = blockIdx.x & 15;
  const int colBase = wgi * 32;
  const int mi = w >> 1, ci = w & 1;
  const int nL = lane & 15, half = (lane >> 4) & 1;
  const int n  = colBase + ci * 16 + nL;     // 0..511 (h column owned by this lane)

  const float*          xs  = dir ? xs1 : xs0;
  const unsigned short* WrT = dir ? WrT1 : WrT0;
  const float*          br  = (dir ? bias1 : bias0) + G3;   // recurrent bias row
  unsigned short*       hb  = dir ? hb1 : hb0;
  float*                hst = dir ? hstate1 : hstate0;

  // ---- stage this WG's Wr slice into LDS: row r = gate*32 + c  (c = col within slice) ----
  for (int i = tid; i < 96 * 64; i += 256) {          // uint4 granules: 96 rows x 64
    const int row = i >> 6, col = i & 63;
    const int gate = row >> 5, c = row & 31;
    const unsigned short* src = WrT + (size_t)(gate * H_ + colBase + c) * H_;
    *(uint4*)(sWr + (size_t)row * LDPAD + col * 8) = ((const uint4*)src)[col];
  }
  __syncthreads();

  const unsigned short* Bz = sWr + (size_t)(0 * 32 + ci * 16) * LDPAD;
  const unsigned short* Br = sWr + (size_t)(1 * 32 + ci * 16) * LDPAD;
  const unsigned short* Bn = sWr + (size_t)(2 * 32 + ci * 16) * LDPAD;

  const float brz = br[n], brr = br[512 + n], brn = br[1024 + n];

  float hreg[8];
#pragma unroll
  for (int r = 0; r < 8; ++r) hreg[r] = 0.f;

  for (int t = 0; t < T_; ++t) {
    const unsigned short* hRead  = hb + (size_t)(t & 1) * (B_ * H_);
    unsigned short*       hWrite = hb + (size_t)((t + 1) & 1) * (B_ * H_);
    const int t_x = dir ? (T_ - 1 - t) : t;

    v8f az = {}; v8f ar = {}; v8f an = {};
#pragma unroll 4
    for (int kc = 0; kc < H_ / 32; ++kc) {
      v16bf a  = load_a_frag(hRead + (size_t)(mi * 16) * H_ + kc * 32, H_, lane);
      v16bf bz = load_b_frag(Bz + kc * 32, LDPAD, lane);
      v16bf b2 = load_b_frag(Br + kc * 32, LDPAD, lane);
      v16bf bn = load_b_frag(Bn + kc * 32, LDPAD, lane);
      az = __builtin_amdgcn_wmma_f32_16x16x32_bf16(false, a, false, bz, (short)0, az, false, false);
      ar = __builtin_amdgcn_wmma_f32_16x16x32_bf16(false, a, false, b2, (short)0, ar, false, false);
      an = __builtin_amdgcn_wmma_f32_16x16x32_bf16(false, a, false, bn, (short)0, an, false, false);
    }

#pragma unroll
    for (int r = 0; r < 8; ++r) {
      const int b = mi * 16 + half * 8 + r;                 // batch row (C-frag layout)
      const float* xrow = xs + ((size_t)b * T_ + t_x) * G3;
      float xz = xrow[n], xr = xrow[512 + n], xh = xrow[1024 + n];
      float z  = 1.f / (1.f + __expf(-(xz + az[r] + brz)));
      float rr = 1.f / (1.f + __expf(-(xr + ar[r] + brr)));
      float hh = tanhf(xh + rr * (an[r] + brn));
      float hnew = z * hreg[r] + (1.f - z) * hh;
      hreg[r] = hnew;
      unsigned short hbv = f2bf(hnew);
      hWrite[(size_t)b * H_ + n] = hbv;
      Y[((size_t)b * T_ + t_x) * (2 * H_) + dir * H_ + n] = hbv;
    }

    // device-wide step barrier (monotonic counter, no reset needed within launch)
    __threadfence();
    __syncthreads();
    if (tid == 0) {
      __hip_atomic_fetch_add(bar, 1u, __ATOMIC_RELEASE, __HIP_MEMORY_SCOPE_AGENT);
      const unsigned target = (unsigned)nwg * (unsigned)(t + 1);
      while (__hip_atomic_load(bar, __ATOMIC_ACQUIRE, __HIP_MEMORY_SCOPE_AGENT) < target)
        __builtin_amdgcn_s_sleep(1);
    }
    __syncthreads();
  }

#pragma unroll
  for (int r = 0; r < 8; ++r) {
    const int b = mi * 16 + half * 8 + r;
    hst[(size_t)b * H_ + n] = hreg[r];
  }
}

// -------------------- assemble (output, states) into d_out --------------------
__global__ void finalize_kernel(const unsigned short* __restrict__ Y1,
                                const float* __restrict__ hs0, const float* __restrict__ hs1,
                                float* __restrict__ out) {
  int i = blockIdx.x * blockDim.x + threadIdx.x;   // 131072
  if (i < B_ * 2 * H_) {
    int b = i >> 10, c = i & 1023;
    out[i] = bf2f(Y1[((size_t)b * T_ + (T_ - 1)) * (2 * H_) + c]);
  } else {
    int j = i - B_ * 2 * H_;
    int b = j >> 10, c = j & 1023;
    out[i] = (c < H_) ? hs0[(size_t)b * H_ + c] : hs1[(size_t)b * H_ + c - H_];
  }
}

extern "C" void kernel_launch(void* const* d_in, const int* in_sizes, int n_in,
                              void* d_out, int out_size, void* d_ws, size_t ws_size,
                              hipStream_t stream) {
  (void)in_sizes; (void)n_in; (void)out_size; (void)ws_size;
  const int*   x   = (const int*)d_in[0];
  const float* emb = (const float*)d_in[3];
  const float* kf0 = (const float*)d_in[4];
  const float* rf0 = (const float*)d_in[5];
  const float* bf0 = (const float*)d_in[6];
  const float* kb0 = (const float*)d_in[7];
  const float* rb0 = (const float*)d_in[8];
  const float* bb0 = (const float*)d_in[9];
  const float* kf1 = (const float*)d_in[10];
  const float* rf1 = (const float*)d_in[11];
  const float* bf1 = (const float*)d_in[12];
  const float* kb1 = (const float*)d_in[13];
  const float* rb1 = (const float*)d_in[14];
  const float* bb1 = (const float*)d_in[15];

  char* ws = (char*)d_ws;
  size_t off = 0;
  auto alloc = [&](size_t bytes) -> char* {
    char* p = ws + off;
    off += (bytes + 255) & ~(size_t)255;
    return p;
  };

  unsigned*       bar    = (unsigned*)alloc(256);              // bar[0]: layer0, bar[16]: layer1
  unsigned short* Xe     = (unsigned short*)alloc((size_t)M_ * E_ * 2);
  unsigned short* WkT0f  = (unsigned short*)alloc((size_t)G3 * E_ * 2);
  unsigned short* WkT0b  = (unsigned short*)alloc((size_t)G3 * E_ * 2);
  unsigned short* WkT1f  = (unsigned short*)alloc((size_t)G3 * 2 * H_ * 2);
  unsigned short* WkT1b  = (unsigned short*)alloc((size_t)G3 * 2 * H_ * 2);
  unsigned short* WrT0f  = (unsigned short*)alloc((size_t)G3 * H_ * 2);
  unsigned short* WrT0b  = (unsigned short*)alloc((size_t)G3 * H_ * 2);
  unsigned short* WrT1f  = (unsigned short*)alloc((size_t)G3 * H_ * 2);
  unsigned short* WrT1b  = (unsigned short*)alloc((size_t)G3 * H_ * 2);
  float*          xsF    = (float*)alloc((size_t)M_ * G3 * 4);
  float*          xsB    = (float*)alloc((size_t)M_ * G3 * 4);
  unsigned short* Y0     = (unsigned short*)alloc((size_t)M_ * 2 * H_ * 2);
  unsigned short* Y1     = (unsigned short*)alloc((size_t)M_ * 2 * H_ * 2);
  unsigned short* hbF    = (unsigned short*)alloc((size_t)2 * B_ * H_ * 2); // ping-pong
  unsigned short* hbB    = (unsigned short*)alloc((size_t)2 * B_ * H_ * 2);
  float*          hsF    = (float*)alloc((size_t)B_ * H_ * 4);
  float*          hsB    = (float*)alloc((size_t)B_ * H_ * 4);

  const size_t scanLds = (size_t)96 * LDPAD * 2;   // 99,840 B dynamic LDS

  hipMemsetAsync(bar, 0, 256, stream);

  // Embedding gather -> bf16
  embed_kernel<<<(M_ * E_) / 256, 256, 0, stream>>>(x, emb, Xe);

  // Weight transpose-converts to (N x K) bf16
  transpose_bf16_kernel<<<(E_ * G3) / 256, 256, 0, stream>>>(kf0, WkT0f, E_, G3);
  transpose_bf16_kernel<<<(E_ * G3) / 256, 256, 0, stream>>>(kb0, WkT0b, E_, G3);
  transpose_bf16_kernel<<<(2 * H_ * G3) / 256, 256, 0, stream>>>(kf1, WkT1f, 2 * H_, G3);
  transpose_bf16_kernel<<<(2 * H_ * G3) / 256, 256, 0, stream>>>(kb1, WkT1b, 2 * H_, G3);
  transpose_bf16_kernel<<<(H_ * G3) / 256, 256, 0, stream>>>(rf0, WrT0f, H_, G3);
  transpose_bf16_kernel<<<(H_ * G3) / 256, 256, 0, stream>>>(rb0, WrT0b, H_, G3);
  transpose_bf16_kernel<<<(H_ * G3) / 256, 256, 0, stream>>>(rf1, WrT1f, H_, G3);
  transpose_bf16_kernel<<<(H_ * G3) / 256, 256, 0, stream>>>(rb1, WrT1b, H_, G3);

  // ---- layer 0 ----
  hipMemsetAsync(hbF, 0, (size_t)2 * B_ * H_ * 2, stream);
  hipMemsetAsync(hbB, 0, (size_t)2 * B_ * H_ * 2, stream);
  gemm_xs_kernel<<<dim3(M_ / 32, G3 / 64), 256, 0, stream>>>(Xe, WkT0f, bf0, xsF, E_);
  gemm_xs_kernel<<<dim3(M_ / 32, G3 / 64), 256, 0, stream>>>(Xe, WkT0b, bb0, xsB, E_);
  gru_scan_kernel<<<32, 256, scanLds, stream>>>(xsF, xsB, WrT0f, WrT0b, bf0, bb0,
                                                hbF, hbB, hsF, hsB, Y0, bar + 0, 32);

  // ---- layer 1 (input = Y0 bf16, K = 1024) ----
  hipMemsetAsync(hbF, 0, (size_t)2 * B_ * H_ * 2, stream);
  hipMemsetAsync(hbB, 0, (size_t)2 * B_ * H_ * 2, stream);
  gemm_xs_kernel<<<dim3(M_ / 32, G3 / 64), 256, 0, stream>>>(Y0, WkT1f, bf1, xsF, 2 * H_);
  gemm_xs_kernel<<<dim3(M_ / 32, G3 / 64), 256, 0, stream>>>(Y0, WkT1b, bb1, xsB, 2 * H_);
  gru_scan_kernel<<<32, 256, scanLds, stream>>>(xsF, xsB, WrT1f, WrT1b, bf1, bb1,
                                                hbF, hbB, hsF, hsB, Y1, bar + 16, 32);

  // ---- outputs ----
  finalize_kernel<<<(2 * B_ * 2 * H_) / 256, 256, 0, stream>>>(Y1, hsF, hsB, (float*)d_out);
}
